// TransformerBlock_79645873537044
// MI455X (gfx1250) — matrix operand order repaired and checked by reading, not compiled, not run on hardware
//
#include <hip/hip_runtime.h>
#include <hip/hip_bf16.h>
#include <math.h>

// ---------------------------------------------------------------------------
// Point-transformer block on MI455X (gfx1250, wave32).
// Dense GEMMs on the matrix pipe via v_wmma_f32_16x16x32_bf16 (f32 accum).
// Weights are prepacked ONCE into bf16 in the per-lane WMMA B-fragment layout
// so every wave fetches its B fragment with two global_load_b128 (32B/lane).
// A fragments use float4 loads (two contiguous 8-float runs per K-chunk).
// Problem is bandwidth-bound (~30-40MB @ 23.3 TB/s); bf16 halves weight bytes.
// ---------------------------------------------------------------------------

#define B_SZ   4
#define NPTS   4096
#define CDIM   64
#define KNN    16
#define GD     8
#define MROWS  (B_SZ * NPTS)   // 16384

typedef __attribute__((ext_vector_type(16))) __bf16 v16bf;
typedef __attribute__((ext_vector_type(8)))  float  v8f;

__device__ __forceinline__ v8f wmma_bf16(v16bf a, v16bf b, v8f c) {
  // (neg_a, A, neg_b, B, c_mod, C, reuse_a, reuse_b)
  return __builtin_amdgcn_wmma_f32_16x16x32_bf16(false, a, false, b, (short)0, c,
                                                 false, false);
}

// A fragment: 16x32 bf16 slice (K-chunk kc) of a row-major f32 row (64 cols).
// Lane L owns row M = L&15; K = 32*kc + (j<4?0:16) + (L>=16?8:0) + 2*(j&3) (+1).
// Per chunk this is two contiguous 8-float runs -> 4x float4 loads.
__device__ __forceinline__ v16bf load_a_row(const float* __restrict__ row, int kc,
                                            int lane) {
  int base = 32 * kc + ((lane >= 16) ? 8 : 0);
  float4 x0 = *(const float4*)(row + base);
  float4 x1 = *(const float4*)(row + base + 4);
  float4 x2 = *(const float4*)(row + base + 16);
  float4 x3 = *(const float4*)(row + base + 20);
  v16bf a;
  a[0]  = (__bf16)x0.x; a[1]  = (__bf16)x0.y; a[2]  = (__bf16)x0.z; a[3]  = (__bf16)x0.w;
  a[4]  = (__bf16)x1.x; a[5]  = (__bf16)x1.y; a[6]  = (__bf16)x1.z; a[7]  = (__bf16)x1.w;
  a[8]  = (__bf16)x2.x; a[9]  = (__bf16)x2.y; a[10] = (__bf16)x2.z; a[11] = (__bf16)x2.w;
  a[12] = (__bf16)x3.x; a[13] = (__bf16)x3.y; a[14] = (__bf16)x3.z; a[15] = (__bf16)x3.w;
  return a;
}

// Prepacked B fragment: contiguous 16 bf16 (32B) per (n0-tile, kchunk, lane).
__device__ __forceinline__ v16bf load_b_packed(const __bf16* __restrict__ P, int n0t,
                                               int kc, int lane) {
  return *(const v16bf*)(P + ((((n0t * 2) + kc) * 32 + lane) << 4));
}

// ---------------------------------------------------------------------------
// Prepack a 64x64 f32 weight (row-major, [in][out]) into bf16 B fragments.
// out[t], t = ((n0t*2+kc)*32+lane)*16 + e, e = 2j+u:
//   N = n0t*16 + (lane&15); K = 32*kc + (j<4?0:16) + (lane>=16?8:0) + 2*(j&3) + u.
// ---------------------------------------------------------------------------
__global__ void wpack64_kernel(const float* __restrict__ W, __bf16* __restrict__ out) {
  int t = blockIdx.x * blockDim.x + threadIdx.x;      // 0..4095
  if (t >= 4096) return;
  int e = t & 15, lane = (t >> 4) & 31, kc = (t >> 9) & 1, n0t = t >> 10;
  int j = e >> 1, u = e & 1;
  int N = n0t * 16 + (lane & 15);
  int K = 32 * kc + ((j < 4) ? 0 : 16) + ((lane >= 16) ? 8 : 0) + 2 * (j & 3) + u;
  out[t] = (__bf16)W[K * 64 + N];
}

// Prepack 64x8 we_w1, zero-padded to 16 output columns (single n0 tile).
__global__ void wpack_we1_kernel(const float* __restrict__ W, __bf16* __restrict__ out) {
  int t = blockIdx.x * blockDim.x + threadIdx.x;      // 0..1023
  if (t >= 1024) return;
  int e = t & 15, lane = (t >> 4) & 31, kc = (t >> 9) & 1;
  int j = e >> 1, u = e & 1;
  int N = lane & 15;
  int K = 32 * kc + ((j < 4) ? 0 : 16) + ((lane >= 16) ? 8 : 0) + 2 * (j & 3) + u;
  out[t] = (N < GD) ? (__bf16)W[K * GD + N] : (__bf16)0.f;
}

// ---------------------------------------------------------------------------
// LayerNorm over [c,n] per batch; optional fused add of att ([b,n,c], transposed)
// producing fea2 ([b,c,n]); writes normalized transposed result t [b,n,c].
// ---------------------------------------------------------------------------
__global__ void ln_kernel(const float* __restrict__ fea, const float* __restrict__ att,
                          float* __restrict__ fea2, float* __restrict__ t) {
  const int total = CDIM * NPTS;
  int bi = blockIdx.x;
  const float* fb = fea + (size_t)bi * total;
  const float* ab = att ? att + (size_t)bi * NPTS * CDIM : nullptr;
  float* f2b = fea2 ? fea2 + (size_t)bi * total : nullptr;
  float* tb = t + (size_t)bi * NPTS * CDIM;

  __shared__ float ssum[512];
  __shared__ float ssq[512];
  float s = 0.f, sq = 0.f;
  for (int i = threadIdx.x; i < total; i += blockDim.x) {
    int ch = i >> 12, pos = i & (NPTS - 1);     // i = ch*NPTS + pos
    float x = fb[i];
    if (ab) x += ab[(size_t)pos * CDIM + ch];
    if (f2b) f2b[i] = x;
    s += x; sq += x * x;
  }
  ssum[threadIdx.x] = s; ssq[threadIdx.x] = sq;
  __syncthreads();
  for (int off = blockDim.x >> 1; off > 0; off >>= 1) {
    if ((int)threadIdx.x < off) {
      ssum[threadIdx.x] += ssum[threadIdx.x + off];
      ssq[threadIdx.x]  += ssq[threadIdx.x + off];
    }
    __syncthreads();
  }
  float mean = ssum[0] / (float)total;
  float var  = ssq[0] / (float)total - mean * mean;
  float inv  = rsqrtf(var + 1e-5f);
  for (int i = threadIdx.x; i < total; i += blockDim.x) {
    int ch = i >> 12, pos = i & (NPTS - 1);
    float x = f2b ? f2b[i] : fb[i];
    tb[(size_t)pos * CDIM + ch] = (x - mean) * inv;
  }
}

// ---------------------------------------------------------------------------
// out[M x 64] = act(A[M x 64] @ Wpacked + bias). One wave per 16-row tile,
// 8 x v_wmma_f32_16x16x32_bf16 per tile; B fragments from prepacked bf16.
// ---------------------------------------------------------------------------
__global__ void gemm64_kernel(const float* __restrict__ A, const __bf16* __restrict__ Wp,
                              const float* __restrict__ bias, float* __restrict__ out,
                              int M, int relu) {
  int lane = threadIdx.x & 31;
  int wv = threadIdx.x >> 5;
  int tile = blockIdx.x * (blockDim.x >> 5) + wv;
  if (tile * 16 >= M) return;                 // wave-uniform

  const float* arow = A + (size_t)(tile * 16 + (lane & 15)) * CDIM;
  v16bf a0 = load_a_row(arow, 0, lane);
  v16bf a1 = load_a_row(arow, 1, lane);

#pragma unroll
  for (int n0 = 0; n0 < 64; n0 += 16) {
    v8f c = {};
    c = wmma_bf16(a0, load_b_packed(Wp, n0 >> 4, 0, lane), c);
    c = wmma_bf16(a1, load_b_packed(Wp, n0 >> 4, 1, lane), c);
    int col = n0 + (lane & 15);
    float bs = bias ? bias[col] : 0.f;
#pragma unroll
    for (int r = 0; r < 8; ++r) {
      int row = tile * 16 + ((lane < 16) ? r : r + 8);
      float v = c[r] + bs;
      if (relu) v = fmaxf(v, 0.f);
      out[(size_t)row * CDIM + col] = v;
    }
  }
}

// ---------------------------------------------------------------------------
// kNN (k=16) brute force: batch xyz staged in LDS (48KB), one thread per query,
// insertion-sorted best-16. d2 uses the same |x|^2+|y|^2-2<x,y> form as the ref.
// ---------------------------------------------------------------------------
__global__ void knn_kernel(const float* __restrict__ xyz, int* __restrict__ idx) {
  __shared__ float sx[NPTS], sy[NPTS], sz[NPTS];
  int bi = blockIdx.y;
  const float* xb = xyz + (size_t)bi * NPTS * 3;
  for (int j = threadIdx.x; j < NPTS; j += blockDim.x) {
    sx[j] = xb[j * 3 + 0]; sy[j] = xb[j * 3 + 1]; sz[j] = xb[j * 3 + 2];
  }
  __syncthreads();

  int q = blockIdx.x * blockDim.x + threadIdx.x;
  float qx = sx[q], qy = sy[q], qz = sz[q];
  float qs = qx * qx + qy * qy + qz * qz;

  float bd[KNN]; int bix[KNN];
#pragma unroll
  for (int t = 0; t < KNN; ++t) { bd[t] = 3.4e38f; bix[t] = 0; }

  for (int j = 0; j < NPTS; ++j) {
    float js = sx[j] * sx[j] + sy[j] * sy[j] + sz[j] * sz[j];
    float d2 = qs + js - 2.f * (qx * sx[j] + qy * sy[j] + qz * sz[j]);
    if (d2 < bd[KNN - 1]) {
      int p = KNN - 1;
      while (p > 0 && d2 < bd[p - 1]) {
        bd[p] = bd[p - 1]; bix[p] = bix[p - 1]; --p;
      }
      bd[p] = d2; bix[p] = j;
    }
  }
  int* op = idx + ((size_t)bi * NPTS + q) * KNN;
#pragma unroll
  for (int t = 0; t < KNN; ++t) op[t] = bix[t];
}

// ---------------------------------------------------------------------------
// First conv of center-pos-encode: h = relu((x3 @ w1) * g/sqrt(1+eps) + b)
// ---------------------------------------------------------------------------
__global__ void enc1_kernel(const float* __restrict__ x3, const float* __restrict__ w1,
                            const float* __restrict__ g, const float* __restrict__ b,
                            float* __restrict__ h) {
  int t = blockIdx.x * blockDim.x + threadIdx.x;
  if (t >= MROWS * CDIM) return;
  int m = t >> 6, o = t & 63;
  float gs = g[o] * rsqrtf(1.f + 1e-5f);
  float acc = x3[m * 3 + 0] * w1[o] + x3[m * 3 + 1] * w1[64 + o] +
              x3[m * 3 + 2] * w1[128 + o];
  h[t] = fmaxf(acc * gs + b[o], 0.f);
}

// ---------------------------------------------------------------------------
// Fused neighborhood attention: one wave32 per point.
//   pe conv1 computed straight into A-fragment layout (registers),
//   we-encode via 2 WMMA (prepacked padded B), softmax over 16 neighbors,
//   pe conv2 via 8 WMMA (prepacked B), gathered V add, shfl_xor(16) reduce.
// ---------------------------------------------------------------------------
struct __align__(16) AttnLds {
  int   nidx[KNN];        // 64B
  float dp[KNN][3];       // 192B
  float qc[CDIM];         // 256B (16B aligned)
  float wa[KNN][GD];
  float wb[KNN][GD];
};

__global__ void attn_kernel(const int* __restrict__ idx, const float* __restrict__ xyz,
                            const float* __restrict__ xq, const float* __restrict__ xk,
                            const float* __restrict__ xv, const float* __restrict__ cpe,
                            const float* __restrict__ pe_w1, const float* __restrict__ pe_g,
                            const float* __restrict__ pe_b,
                            const __bf16* __restrict__ pe2_p,   // prepacked pe_w2
                            const float* __restrict__ pe_b2,
                            const __bf16* __restrict__ we1_p,   // prepacked we_w1 (padded)
                            const float* __restrict__ we_g, const float* __restrict__ we_b,
                            const float* __restrict__ we_w2, const float* __restrict__ we_b2,
                            float* __restrict__ att) {
  __shared__ AttnLds ws[4];
  int lane = threadIdx.x & 31;
  int wv = threadIdx.x >> 5;
  AttnLds& S = ws[wv];
  int p = blockIdx.x * 4 + wv;      // 0..16383, exact grid
  int bi = p >> 12;
  int pi = p & (NPTS - 1);
  const float bnr = rsqrtf(1.f + 1e-5f);

  // ---- stage indices, relative positions, (cpe - xq) row; prefetch K/V rows ----
  if (lane < KNN) {
    int nb = idx[(size_t)p * KNN + lane];
    S.nidx[lane] = nb;
    const float* nx = xyz + ((size_t)bi * NPTS + nb) * 3;
    const float* cx = xyz + ((size_t)bi * NPTS + pi) * 3;
    S.dp[lane][0] = nx[0] - cx[0];
    S.dp[lane][1] = nx[1] - cx[1];
    S.dp[lane][2] = nx[2] - cx[2];
    // gfx1250 global_prefetch: pull gathered neighbor rows toward the WGP early
    __builtin_prefetch(xk + ((size_t)bi * NPTS + nb) * CDIM, 0, 3);
    __builtin_prefetch(xv + ((size_t)bi * NPTS + nb) * CDIM, 0, 3);
  }
  {
    const float* cp = cpe + (size_t)p * CDIM;
    const float* qp = xq + (size_t)p * CDIM;
    S.qc[lane]      = cp[lane]      - qp[lane];
    S.qc[lane + 32] = cp[lane + 32] - qp[lane + 32];
  }
  __syncthreads();

  int m  = lane & 15;
  int hi = (lane >= 16) ? 8 : 0;

  // ---- pe conv1 directly into A fragments ----
  float d0 = S.dp[m][0], d1 = S.dp[m][1], d2 = S.dp[m][2];
  v16bf aPe[2];
#pragma unroll
  for (int kc = 0; kc < 2; ++kc) {
#pragma unroll
    for (int j = 0; j < 8; ++j) {
      int k0 = 32 * kc + ((j < 4) ? 0 : 16) + hi + 2 * (j & 3);
#pragma unroll
      for (int u = 0; u < 2; ++u) {
        int k = k0 + u;
        float h = d0 * pe_w1[k] + d1 * pe_w1[64 + k] + d2 * pe_w1[128 + k];
        h = fmaxf(h * (pe_g[k] * bnr) + pe_b[k], 0.f);
        aPe[kc][2 * j + u] = (__bf16)h;
      }
    }
  }

  // ---- r_qk = nk - xq + cpe into A fragments (gathered rows, float4) ----
  const float* krow = xk + ((size_t)bi * NPTS + S.nidx[m]) * CDIM;
  v16bf aR[2];
#pragma unroll
  for (int kc = 0; kc < 2; ++kc) {
    int base = 32 * kc + hi;
#pragma unroll
    for (int half = 0; half < 2; ++half) {
      int o = base + half * 16;
      float4 x0 = *(const float4*)(krow + o);
      float4 x1 = *(const float4*)(krow + o + 4);
      float4 q0 = *(const float4*)(&S.qc[o]);
      float4 q1 = *(const float4*)(&S.qc[o + 4]);
      aR[kc][8 * half + 0] = (__bf16)(x0.x + q0.x);
      aR[kc][8 * half + 1] = (__bf16)(x0.y + q0.y);
      aR[kc][8 * half + 2] = (__bf16)(x0.z + q0.z);
      aR[kc][8 * half + 3] = (__bf16)(x0.w + q0.w);
      aR[kc][8 * half + 4] = (__bf16)(x1.x + q1.x);
      aR[kc][8 * half + 5] = (__bf16)(x1.y + q1.y);
      aR[kc][8 * half + 6] = (__bf16)(x1.z + q1.z);
      aR[kc][8 * half + 7] = (__bf16)(x1.w + q1.w);
    }
  }

  // ---- weight-encode conv1: r_qk(16x64) @ we_w1(64x8 padded) ----
  v8f cw = {};
  cw = wmma_bf16(aR[0], load_b_packed(we1_p, 0, 0, lane), cw);
  cw = wmma_bf16(aR[1], load_b_packed(we1_p, 0, 1, lane), cw);
  {
    int ncol = lane & 15;
    if (ncol < GD) {
      float gsv = we_g[ncol] * bnr, bbv = we_b[ncol];
#pragma unroll
      for (int r = 0; r < 8; ++r) {
        int row = (lane < 16) ? r : r + 8;
        S.wa[row][ncol] = fmaxf(cw[r] * gsv + bbv, 0.f);
      }
    }
  }
  __syncthreads();

  // ---- weight-encode conv2 (8x8) ----
#pragma unroll
  for (int e = 0; e < 4; ++e) {
    int eid = lane * 4 + e;       // 0..127
    int row = eid >> 3, go = eid & 7;
    float acc = we_b2[go];
#pragma unroll
    for (int gi = 0; gi < 8; ++gi) acc += S.wa[row][gi] * we_w2[gi * 8 + go];
    S.wb[row][go] = acc;
  }
  __syncthreads();

  // ---- softmax over 16 neighbors, per group ----
  if (lane < GD) {
    float mx = -1e30f;
    for (int r = 0; r < KNN; ++r) mx = fmaxf(mx, S.wb[r][lane]);
    float ex[KNN], sum = 0.f;
    for (int r = 0; r < KNN; ++r) { ex[r] = __expf(S.wb[r][lane] - mx); sum += ex[r]; }
    float inv = 1.f / sum;
    for (int r = 0; r < KNN; ++r) S.wb[r][lane] = ex[r] * inv;
  }
  __syncthreads();

  // ---- pe conv2 + gathered V + weighted reduction over neighbors ----
#pragma unroll
  for (int n0 = 0; n0 < 64; n0 += 16) {
    v8f c = {};
    c = wmma_bf16(aPe[0], load_b_packed(pe2_p, n0 >> 4, 0, lane), c);
    c = wmma_bf16(aPe[1], load_b_packed(pe2_p, n0 >> 4, 1, lane), c);
    int ncol = n0 + (lane & 15);
    float pb = pe_b2[ncol];
    int go = ncol >> 3;
    float partial = 0.f;
#pragma unroll
    for (int r = 0; r < 8; ++r) {
      int row = (lane < 16) ? r : r + 8;
      float vv = c[r] + pb + xv[((size_t)bi * NPTS + S.nidx[row]) * CDIM + ncol];
      partial += vv * S.wb[row][go];
    }
    partial += __shfl_xor(partial, 16, 32);    // combine rows 0-7 with 8-15
    if (lane < 16) att[(size_t)p * CDIM + ncol] = partial;
  }
}

// ---------------------------------------------------------------------------
// Final: out = [xyz flat | fea3 [b,c,n]] with fea3 = fea2 + o^T.
// ---------------------------------------------------------------------------
__global__ void final_kernel(const float* __restrict__ xyz, const float* __restrict__ fea2,
                             const float* __restrict__ o, float* __restrict__ out) {
  const int XYZN = B_SZ * NPTS * 3;
  int t = blockIdx.x * blockDim.x + threadIdx.x;
  if (t < XYZN) out[t] = xyz[t];
  if (t < B_SZ * CDIM * NPTS) {
    int bi = t >> 18;                 // CDIM*NPTS = 262144
    int rem = t & (262144 - 1);
    int ch = rem >> 12, pos = rem & (NPTS - 1);
    out[XYZN + t] = fea2[t] + o[((size_t)bi * NPTS + pos) * CDIM + ch];
  }
}

// ---------------------------------------------------------------------------
extern "C" void kernel_launch(void* const* d_in, const int* in_sizes, int n_in,
                              void* d_out, int out_size, void* d_ws, size_t ws_size,
                              hipStream_t stream) {
  const float* xyz   = (const float*)d_in[0];
  const float* fea   = (const float*)d_in[1];
  const float* Wq    = (const float*)d_in[2];
  const float* bq    = (const float*)d_in[3];
  const float* Wk    = (const float*)d_in[4];
  const float* bk    = (const float*)d_in[5];
  const float* Wv    = (const float*)d_in[6];
  const float* bv    = (const float*)d_in[7];
  const float* cpe_w1 = (const float*)d_in[8];
  const float* cpe_g  = (const float*)d_in[9];
  const float* cpe_b  = (const float*)d_in[10];
  const float* cpe_w2 = (const float*)d_in[11];
  const float* cpe_b2 = (const float*)d_in[12];
  const float* pe_w1  = (const float*)d_in[13];
  const float* pe_g   = (const float*)d_in[14];
  const float* pe_b   = (const float*)d_in[15];
  const float* pe_w2  = (const float*)d_in[16];
  const float* pe_b2  = (const float*)d_in[17];
  const float* we_w1  = (const float*)d_in[18];
  const float* we_g   = (const float*)d_in[19];
  const float* we_b   = (const float*)d_in[20];
  const float* we_w2  = (const float*)d_in[21];
  const float* we_b2  = (const float*)d_in[22];
  const float* fc1_w  = (const float*)d_in[23];
  const float* fc1_b  = (const float*)d_in[24];
  const float* fc2_w  = (const float*)d_in[25];
  const float* fc2_b  = (const float*)d_in[26];

  float* ws = (float*)d_ws;
  const size_t S = (size_t)MROWS * CDIM;     // 1,048,576 floats
  float* ft   = ws + 0 * S;                  // LN1 output [b,n,c]
  float* xq   = ws + 1 * S;
  float* xk   = ws + 2 * S;
  float* xv   = ws + 3 * S;
  float* cpe  = ws + 4 * S;
  float* cpeh = ws + 5 * S;                  // cpe hidden; reused as att
  float* fea2 = ws + 6 * S;
  int*   idxb = (int*)(ws + 7 * S);          // [b,n,16] = 262144 ints
  // prepacked bf16 weights (32B-aligned: byte offset (7S+262144)*4 % 32 == 0)
  __bf16* pk  = (__bf16*)(ws + 7 * S + (size_t)MROWS * KNN);
  __bf16* Wq_p   = pk + 0 * 4096;
  __bf16* Wk_p   = pk + 1 * 4096;
  __bf16* Wv_p   = pk + 2 * 4096;
  __bf16* cpe2_p = pk + 3 * 4096;
  __bf16* fc1_p  = pk + 4 * 4096;
  __bf16* fc2_p  = pk + 5 * 4096;
  __bf16* pe2_p  = pk + 6 * 4096;
  __bf16* we1_p  = pk + 7 * 4096;            // 1024 used

  float* att  = cpeh;                        // reuse (cpeh dead after cpe GEMM)
  float* t2   = ft;                          // reuse (ft dead after qkv)
  float* hmlp = xq;                          // reuse (xq dead after attn)
  float* omlp = xk;                          // reuse (xk dead after attn)

  const int GEMM_BLK = 128;                  // 4 waves
  const int GEMM_GRID = MROWS / 16 / 4;      // 256

  // prepack weights into WMMA B-fragment bf16 layout (once per launch)
  wpack64_kernel<<<8, 512, 0, stream>>>(Wq, Wq_p);
  wpack64_kernel<<<8, 512, 0, stream>>>(Wk, Wk_p);
  wpack64_kernel<<<8, 512, 0, stream>>>(Wv, Wv_p);
  wpack64_kernel<<<8, 512, 0, stream>>>(cpe_w2, cpe2_p);
  wpack64_kernel<<<8, 512, 0, stream>>>(fc1_w, fc1_p);
  wpack64_kernel<<<8, 512, 0, stream>>>(fc2_w, fc2_p);
  wpack64_kernel<<<8, 512, 0, stream>>>(pe_w2, pe2_p);
  wpack_we1_kernel<<<2, 512, 0, stream>>>(we_w1, we1_p);

  // layernorm1 -> ft
  ln_kernel<<<B_SZ, 512, 0, stream>>>(fea, nullptr, nullptr, ft);
  // q/k/v projections (WMMA)
  gemm64_kernel<<<GEMM_GRID, GEMM_BLK, 0, stream>>>(ft, Wq_p, bq, xq, MROWS, 0);
  gemm64_kernel<<<GEMM_GRID, GEMM_BLK, 0, stream>>>(ft, Wk_p, bk, xk, MROWS, 0);
  gemm64_kernel<<<GEMM_GRID, GEMM_BLK, 0, stream>>>(ft, Wv_p, bv, xv, MROWS, 0);
  // kNN (LDS-staged brute force)
  knn_kernel<<<dim3(NPTS / 256, B_SZ), 256, 0, stream>>>(xyz, idxb);
  // center position encode: scalar conv1 + WMMA conv2
  enc1_kernel<<<(MROWS * CDIM + 255) / 256, 256, 0, stream>>>(xyz, cpe_w1, cpe_g,
                                                              cpe_b, cpeh);
  gemm64_kernel<<<GEMM_GRID, GEMM_BLK, 0, stream>>>(cpeh, cpe2_p, cpe_b2, cpe, MROWS, 0);
  // fused neighborhood attention (10 WMMA per point-wave)
  attn_kernel<<<MROWS / 4, 128, 0, stream>>>(idxb, xyz, xq, xk, xv, cpe,
                                             pe_w1, pe_g, pe_b, pe2_p, pe_b2,
                                             we1_p, we_g, we_b, we_w2, we_b2, att);
  // residual + layernorm2 -> t2, fea2
  ln_kernel<<<B_SZ, 512, 0, stream>>>(fea, att, fea2, t2);
  // MLP (WMMA)
  gemm64_kernel<<<GEMM_GRID, GEMM_BLK, 0, stream>>>(t2, fc1_p, fc1_b, hmlp, MROWS, 1);
  gemm64_kernel<<<GEMM_GRID, GEMM_BLK, 0, stream>>>(hmlp, fc2_p, fc2_b, omlp, MROWS, 0);
  // final residual + transpose + xyz passthrough
  final_kernel<<<(B_SZ * CDIM * NPTS + 255) / 256, 256, 0, stream>>>(xyz, fea2, omlp,
                                                                     (float*)d_out);
}
